// cross_correlation_block_45999099740215
// MI455X (gfx1250) — matrix operand order repaired and checked by reading, not compiled, not run on hardware
//
#include <hip/hip_runtime.h>
#include <hip/hip_bf16.h>
#include <cstdint>
#include <cstddef>

// ---------------------------------------------------------------------------
// Problem constants (from reference)
// ---------------------------------------------------------------------------
static constexpr int BQ  = 16;    // batch
static constexpr int CH  = 512;   // channels
static constexpr int CIq = 256;   // inter channels
static constexpr int NP  = 1024;  // H*W
static constexpr float EPSV = 1e-5f;

// ---------------------------------------------------------------------------
// Types / helpers
// ---------------------------------------------------------------------------
typedef __bf16 bf16_t;
typedef bf16_t v16bf __attribute__((ext_vector_type(16)));
typedef float  v8f   __attribute__((ext_vector_type(8)));

__device__ __forceinline__ bf16_t f2bf(float f) {
  uint32_t u = __builtin_bit_cast(uint32_t, f);
  uint32_t r = u + 0x7fffu + ((u >> 16) & 1u);   // round-to-nearest-even
  uint16_t h = (uint16_t)(r >> 16);
  return __builtin_bit_cast(bf16_t, h);
}
__device__ __forceinline__ float bf2f(bf16_t x) {
  uint16_t h = __builtin_bit_cast(uint16_t, x);
  uint32_t u = ((uint32_t)h) << 16;
  return __builtin_bit_cast(float, u);
}
__device__ __forceinline__ float  ld_elem(const float*  p, size_t i) { return p[i]; }
__device__ __forceinline__ float  ld_elem(const bf16_t* p, size_t i) { return bf2f(p[i]); }
__device__ __forceinline__ void   st_elem(float*  p, size_t i, float v) { p[i] = v; }
__device__ __forceinline__ void   st_elem(bf16_t* p, size_t i, float v) { p[i] = f2bf(v); }

// ---------------------------------------------------------------------------
// 128x128-block bf16 WMMA GEMM:  Cout[m,n] = sum_k A[m,k]*B[k,n] (+bias[m]).
// REQUIRES: M % 128 == 0, N % 128 == 0, K % 32 == 0 (true for every GEMM in
// this problem) -> no bounds checks, no exec-mask branching in the hot loop.
//
// 256 threads = 8 waves (2x4); each wave owns a 64x32 patch = 4x2 accumulator
// tiles -> 8 v_wmma_f32_16x16x32_bf16 per K-step from 6 fragment loads.
// B tile is stored TRANSPOSED in LDS so both fragment layouts are contiguous
// 16-byte runs per lane (merge into ds_load_b128, no v_mov_b16 merges):
//   A frag: k = lh*8 + e (e<8), 16 + lh*8 + (e-8) (e>=8)   [ISA 7.12.2]
//   B frag: k = lh*8 + e (e<8), 16 + lh*8 + (e-8) (e>=8)                                   [ISA 7.12.2]
// Optional fused softmax on B: b = exp(b - smax[col]) * sinv[col].
// ---------------------------------------------------------------------------
template<typename TA, typename TB, typename TOUT, bool SOFTMAX_B, bool HAS_BIAS>
__launch_bounds__(256)
__global__ void gemm128x128_wmma(
    const TA* __restrict__ A, long long a_batch, int a_rs, int a_ks,
    const TB* __restrict__ B, long long b_batch, int b_ks, int b_ns,
    TOUT* __restrict__ Cout,  long long c_batch, int c_ld,
    const float* __restrict__ bias,
    const float* __restrict__ smax, const float* __restrict__ sinv, int stat_stride,
    int M, int N, int K)
{
  constexpr int BM = 128, BN = 128, BK = 32;
  constexpr int LDA = BK + 8;   // 40 elems = 80 B row pitch (16B multiple)
  constexpr int LDB = BK + 8;
  __shared__ alignas(16) bf16_t As [BM * LDA];   // [m][k]
  __shared__ alignas(16) bf16_t BsT[BN * LDB];   // [n][k]  (transposed)

  const int tid  = threadIdx.x;
  const int lane = tid & 31;
  const int wave = tid >> 5;
  const int wm   = (wave >> 2) * 64;     // wave row offset (0 / 64)
  const int wn   = (wave & 3)  * 32;     // wave col offset (0/32/64/96)
  const int lm   = lane & 15;            // lane index within 16
  const int lh   = lane >> 4;            // lane half (0/1)

  const int m0 = blockIdx.y * BM;
  const int n0 = blockIdx.x * BN;
  const int bz = blockIdx.z;

  const TA* Ab = A + (size_t)bz * a_batch;
  const TB* Bb = B + (size_t)bz * b_batch;
  TOUT*     Cb = Cout + (size_t)bz * c_batch;
  const float* smaxb = SOFTMAX_B ? (smax + (size_t)bz * stat_stride) : nullptr;
  const float* sinvb = SOFTMAX_B ? (sinv + (size_t)bz * stat_stride) : nullptr;

  v8f acc[4][2];
  #pragma unroll
  for (int i = 0; i < 4; ++i)
    #pragma unroll
    for (int j = 0; j < 2; ++j)
      #pragma unroll
      for (int e = 0; e < 8; ++e) acc[i][j][e] = 0.0f;

  for (int k0 = 0; k0 < K; k0 += BK) {
    // Hint next K-tile toward L2 (global_prefetch_b8).
    if (k0 + BK < K) {
      __builtin_prefetch(Ab + (size_t)(m0 + (tid >> 1)) * a_rs + (size_t)(k0 + BK) * a_ks, 0, 1);
      __builtin_prefetch(Bb + (size_t)(k0 + BK) * b_ks + (size_t)(n0 + (tid >> 1)) * b_ns, 0, 1);
    }

    // ---- stage A tile: 128 x 32 (4096 elems / 256 threads = 16 each) ----
    #pragma unroll
    for (int it = 0; it < 16; ++it) {
      int e = tid + it * 256;
      int r = e >> 5;            // 0..127
      int k = e & 31;            // 0..31  (fast index -> coalesced when a_ks==1)
      float v = ld_elem(Ab, (size_t)(m0 + r) * a_rs + (size_t)(k0 + k) * a_ks);
      As[r * LDA + k] = f2bf(v);
    }
    // ---- stage B tile transposed: [n][k], 32 x 128 source ----
    #pragma unroll
    for (int it = 0; it < 16; ++it) {
      int e = tid + it * 256;
      int n = e & 127;           // 0..127 (fast index -> coalesced when b_ns==1)
      int k = e >> 7;            // 0..31
      float v = ld_elem(Bb, (size_t)(k0 + k) * b_ks + (size_t)(n0 + n) * b_ns);
      if constexpr (SOFTMAX_B)
        v = __expf(v - smaxb[n0 + n]) * sinvb[n0 + n];
      BsT[n * LDB + k] = f2bf(v);
    }
    __syncthreads();

    // ---- fragments: contiguous 16B runs per lane -> ds_load_b128 ----
    v16bf afr[4], bfr[2];
    #pragma unroll
    for (int i = 0; i < 4; ++i) {
      const bf16_t* pa = &As[(wm + i * 16 + lm) * LDA];
      #pragma unroll
      for (int e = 0; e < 8; ++e) afr[i][e]     = pa[lh * 8 + e];
      #pragma unroll
      for (int e = 0; e < 8; ++e) afr[i][8 + e] = pa[16 + lh * 8 + e];
    }
    #pragma unroll
    for (int j = 0; j < 2; ++j) {
      const bf16_t* pb = &BsT[(wn + j * 16 + lm) * LDB];
      #pragma unroll
      for (int e = 0; e < 8; ++e) bfr[j][e]     = pb[lh * 8 + e];
      #pragma unroll
      for (int e = 0; e < 8; ++e) bfr[j][8 + e] = pb[16 + lh * 8 + e];
    }

    #pragma unroll
    for (int i = 0; i < 4; ++i)
      #pragma unroll
      for (int j = 0; j < 2; ++j)
        acc[i][j] = __builtin_amdgcn_wmma_f32_16x16x32_bf16(
            false, afr[i], false, bfr[j], (short)0, acc[i][j], false, false);

    __syncthreads();
  }

  // ---- store C/D (lane layout: col = lane%16, row = (lane/16)*8 + r) ----
  #pragma unroll
  for (int i = 0; i < 4; ++i) {
    #pragma unroll
    for (int j = 0; j < 2; ++j) {
      int rb = m0 + wm + i * 16;
      int cb = n0 + wn + j * 16 + lm;
      #pragma unroll
      for (int r = 0; r < 8; ++r) {
        int gm = rb + lh * 8 + r;
        float v = acc[i][j][r];
        if constexpr (HAS_BIAS) v += bias[gm];
        st_elem(Cb, (size_t)gm * c_ld + cb, v);
      }
    }
  }
}

// ---------------------------------------------------------------------------
// Softmax stats over rows of scores (wave per row): rmax, rinv = 1/sum(exp)
// rows are (b*NP + n), contiguous length ncols.
// ---------------------------------------------------------------------------
__global__ void row_softmax_stats(const float* __restrict__ S,
                                  float* __restrict__ rmax, float* __restrict__ rinv,
                                  int nrows, int ncols)
{
  int wave = (int)((blockIdx.x * blockDim.x + threadIdx.x) >> 5);
  int lane = threadIdx.x & 31;
  if (wave >= nrows) return;
  const float* row = S + (size_t)wave * ncols;
  float m = -3.4e38f;
  for (int j = lane; j < ncols; j += 32) m = fmaxf(m, row[j]);
  #pragma unroll
  for (int off = 16; off; off >>= 1) m = fmaxf(m, __shfl_xor(m, off, 32));
  float s = 0.0f;
  for (int j = lane; j < ncols; j += 32) s += __expf(row[j] - m);
  #pragma unroll
  for (int off = 16; off; off >>= 1) s += __shfl_xor(s, off, 32);
  if (lane == 0) { rmax[wave] = m; rinv[wave] = 1.0f / s; }
}

// Column stats (thread per column, coalesced across threads).
__global__ void col_softmax_stats(const float* __restrict__ S,
                                  float* __restrict__ cmax, float* __restrict__ cinv,
                                  int nmat, int n)
{
  int g = blockIdx.x * blockDim.x + threadIdx.x;   // b*n + m
  if (g >= nmat * n) return;
  int b = g / n, m = g % n;
  const float* base = S + (size_t)b * n * n + m;
  float mx = -3.4e38f;
  for (int i = 0; i < n; ++i) mx = fmaxf(mx, base[(size_t)i * n]);
  float s = 0.0f;
  for (int i = 0; i < n; ++i) s += __expf(base[(size_t)i * n] - mx);
  cmax[g] = mx; cinv[g] = 1.0f / s;
}

// ---------------------------------------------------------------------------
// BatchNorm (training mode): per-channel mean / rsqrt(var+eps) over (B, N).
// Z layout: [B, C, N]. One block per channel.
// ---------------------------------------------------------------------------
__global__ void bn_reduce(const float* __restrict__ Z,
                          float* __restrict__ mean, float* __restrict__ rstd,
                          int Bn, int Cn, int Nn)
{
  int c = blockIdx.x;
  int t = threadIdx.x;
  float s = 0.0f, s2 = 0.0f;
  for (int b = 0; b < Bn; ++b) {
    const float* p = Z + ((size_t)b * Cn + c) * Nn;
    for (int n = t; n < Nn; n += (int)blockDim.x) { float v = p[n]; s += v; s2 += v * v; }
  }
  __shared__ float sh[256], sh2[256];
  sh[t] = s; sh2[t] = s2;
  __syncthreads();
  for (int o = 128; o; o >>= 1) {
    if (t < o) { sh[t] += sh[t + o]; sh2[t] += sh2[t + o]; }
    __syncthreads();
  }
  if (t == 0) {
    float cnt = (float)Bn * (float)Nn;
    float mu  = sh[0] / cnt;
    float var = sh2[0] / cnt - mu * mu;
    mean[c] = mu;
    rstd[c] = rsqrtf(var + EPSV);
  }
}

// z = (z - mean[c]) * rstd[c] * gamma[c] + beta[c] + residual  (in place)
__global__ void bn_apply(float* __restrict__ Z, const float* __restrict__ X,
                         const float* __restrict__ mean, const float* __restrict__ rstd,
                         const float* __restrict__ gamma, const float* __restrict__ beta,
                         int total, int Nn, int Cn)
{
  int i = blockIdx.x * blockDim.x + threadIdx.x;
  if (i >= total) return;
  int c = (i / Nn) % Cn;
  float z = Z[i];
  Z[i] = (z - mean[c]) * rstd[c] * gamma[c] + beta[c] + X[i];
}

// ---------------------------------------------------------------------------
// Orchestration
// ---------------------------------------------------------------------------
extern "C" void kernel_launch(void* const* d_in, const int* in_sizes, int n_in,
                              void* d_out, int out_size, void* d_ws, size_t ws_size,
                              hipStream_t stream) {
  (void)in_sizes; (void)n_in; (void)out_size; (void)ws_size;

  const float* rgb      = (const float*)d_in[0];   // [B,C,N]
  const float* flow     = (const float*)d_in[1];
  const float* wg_rgb   = (const float*)d_in[2];   // [CI,C]
  const float* bg_rgb   = (const float*)d_in[3];
  const float* wg_flow  = (const float*)d_in[4];
  const float* bg_flow  = (const float*)d_in[5];
  const float* ww_rgb   = (const float*)d_in[6];   // [C,CI]
  const float* bw_rgb   = (const float*)d_in[7];
  const float* g_rgb    = (const float*)d_in[8];
  const float* be_rgb   = (const float*)d_in[9];
  const float* ww_flow  = (const float*)d_in[10];
  const float* bw_flow  = (const float*)d_in[11];
  const float* g_flow   = (const float*)d_in[12];
  const float* be_flow  = (const float*)d_in[13];

  float* out = (float*)d_out;
  float* rgbZ  = out;                                 // [B,C,N]
  float* flowZ = out + (size_t)BQ * CH * NP;

  // ---- workspace carve (deterministic) ----
  char* w = (char*)d_ws;
  auto carve = [&](size_t bytes) {
    void* p = (void*)w;
    w += (bytes + 255) & ~(size_t)255;
    return p;
  };
  bf16_t* reE   = (bf16_t*)carve((size_t)BQ * CIq * NP * sizeof(bf16_t));
  bf16_t* feE   = (bf16_t*)carve((size_t)BQ * CIq * NP * sizeof(bf16_t));
  bf16_t* rgbY  = (bf16_t*)carve((size_t)BQ * CIq * NP * sizeof(bf16_t));
  bf16_t* flowY = (bf16_t*)carve((size_t)BQ * CIq * NP * sizeof(bf16_t));
  float*  rowmax = (float*)carve((size_t)BQ * NP * sizeof(float));
  float*  rowinv = (float*)carve((size_t)BQ * NP * sizeof(float));
  float*  colmax = (float*)carve((size_t)BQ * NP * sizeof(float));
  float*  colinv = (float*)carve((size_t)BQ * NP * sizeof(float));
  float*  meanR  = (float*)carve((size_t)CH * sizeof(float));
  float*  rstdR  = (float*)carve((size_t)CH * sizeof(float));
  float*  meanF  = (float*)carve((size_t)CH * sizeof(float));
  float*  rstdF  = (float*)carve((size_t)CH * sizeof(float));
  float*  scores = (float*)carve((size_t)BQ * NP * NP * sizeof(float));

  const dim3 blk(256);

  // 1) Embeddings:  E[b] = Wg (CI x C) * X[b] (C x N) + bg   -> bf16 [B,CI,N]
  {
    dim3 grid(NP / 128, CIq / 128, BQ);
    gemm128x128_wmma<float, float, bf16_t, false, true><<<grid, blk, 0, stream>>>(
        wg_rgb, 0LL, CH, 1,
        rgb, (long long)CH * NP, NP, 1,
        reE, (long long)CIq * NP, NP,
        bg_rgb, nullptr, nullptr, 0,
        CIq, NP, CH);
    gemm128x128_wmma<float, float, bf16_t, false, true><<<grid, blk, 0, stream>>>(
        wg_flow, 0LL, CH, 1,
        flow, (long long)CH * NP, NP, 1,
        feE, (long long)CIq * NP, NP,
        bg_flow, nullptr, nullptr, 0,
        CIq, NP, CH);
  }

  // 2) Scores:  S[b] (N x N) = reE[b]^T (N x CI) * feE[b] (CI x N)   -> f32
  {
    dim3 grid(NP / 128, NP / 128, BQ);
    gemm128x128_wmma<bf16_t, bf16_t, float, false, false><<<grid, blk, 0, stream>>>(
        reE, (long long)CIq * NP, 1, NP,     // A[n,i] = reE[i,n]  (transposed)
        feE, (long long)CIq * NP, NP, 1,
        scores, (long long)NP * NP, NP,
        nullptr, nullptr, nullptr, 0,
        NP, NP, CIq);
  }

  // 3) Softmax stats (row softmax for rgb attention, column softmax for flow)
  {
    int nrows = BQ * NP;
    row_softmax_stats<<<dim3(nrows / 8), dim3(256), 0, stream>>>(scores, rowmax, rowinv, nrows, NP);
    col_softmax_stats<<<dim3((BQ * NP) / 256), dim3(256), 0, stream>>>(scores, colmax, colinv, BQ, NP);
  }

  // 4) Attention GEMMs with fused softmax on the B (score) operand.
  //    rgb_y[i,n] = sum_m softmax_row(S)[n,m] * feE[i,m]
  //    flow_y[i,m] = sum_n softmax_col(S)[n,m] * reE[i,n]
  {
    dim3 grid(NP / 128, CIq / 128, BQ);
    gemm128x128_wmma<bf16_t, float, bf16_t, true, false><<<grid, blk, 0, stream>>>(
        feE, (long long)CIq * NP, NP, 1,
        scores, (long long)NP * NP, 1, NP,   // B[k=m, col=n] = S[n,m]
        rgbY, (long long)CIq * NP, NP,
        nullptr, rowmax, rowinv, NP,
        CIq, NP, NP);
    gemm128x128_wmma<bf16_t, float, bf16_t, true, false><<<grid, blk, 0, stream>>>(
        reE, (long long)CIq * NP, NP, 1,
        scores, (long long)NP * NP, NP, 1,   // B[k=n, col=m] = S[n,m]
        flowY, (long long)CIq * NP, NP,
        nullptr, colmax, colinv, NP,
        CIq, NP, NP);
  }

  // 5) Output convs into d_out:  Z[b] = Ww (C x CI) * Y[b] (CI x N) + bw
  {
    dim3 grid(NP / 128, CH / 128, BQ);
    gemm128x128_wmma<float, bf16_t, float, false, true><<<grid, blk, 0, stream>>>(
        ww_rgb, 0LL, CIq, 1,
        rgbY, (long long)CIq * NP, NP, 1,
        rgbZ, (long long)CH * NP, NP,
        bw_rgb, nullptr, nullptr, 0,
        CH, NP, CIq);
    gemm128x128_wmma<float, bf16_t, float, false, true><<<grid, blk, 0, stream>>>(
        ww_flow, 0LL, CIq, 1,
        flowY, (long long)CIq * NP, NP, 1,
        flowZ, (long long)CH * NP, NP,
        bw_flow, nullptr, nullptr, 0,
        CH, NP, CIq);
  }

  // 6) BatchNorm (training stats) + residual, in place on d_out.
  {
    bn_reduce<<<dim3(CH), dim3(256), 0, stream>>>(rgbZ,  meanR, rstdR, BQ, CH, NP);
    bn_reduce<<<dim3(CH), dim3(256), 0, stream>>>(flowZ, meanF, rstdF, BQ, CH, NP);
    int total = BQ * CH * NP;
    bn_apply<<<dim3(total / 256), dim3(256), 0, stream>>>(
        rgbZ, rgb, meanR, rstdR, g_rgb, be_rgb, total, NP, CH);
    bn_apply<<<dim3(total / 256), dim3(256), 0, stream>>>(
        flowZ, flow, meanF, rstdF, g_flow, be_flow, total, NP, CH);
  }
}
